// MixtureOfExperts_66571993088379
// MI455X (gfx1250) — matrix operand order repaired and checked by reading, not compiled, not run on hardware
//
#include <hip/hip_runtime.h>
#include <stdint.h>

// ---------------- MoE config (matches reference) ----------------
#define T_TOKENS 8192     // B*S = 2*4096
#define DMODEL   1024
#define DFF      4096
#define NEXP     8

#define BM       64       // tokens per FFN block tile
#define BF       256      // D_FF chunk per outer iteration (16 waves x 16 cols)
#define NWAVES   16       // 512 threads
#define NKT1     (DMODEL/32)   // 32  k-tiles for x@w1
#define NT1      (DFF/16)      // 256 n-tiles of w1
#define NKT2     (DFF/32)      // 128 k-tiles for h@w2
#define NT2      (DMODEL/16)   // 64  n-tiles of w2

typedef __attribute__((ext_vector_type(16))) __bf16 v16bf;
typedef __attribute__((ext_vector_type(8)))  __bf16 v8bf;
typedef __attribute__((ext_vector_type(8)))  float  v8f;
typedef __attribute__((ext_vector_type(4)))  float  v4f;

static __device__ __forceinline__ v16bf cat8(v8bf lo, v8bf hi) {
    return __builtin_shufflevector(lo, hi, 0,1,2,3,4,5,6,7,8,9,10,11,12,13,14,15);
}

// ===================== Router =====================
// One wave per token: logits = x[t]@rw + rb, top-2, 2-way softmax,
// atomic-append (token, gate) into per-expert lists.
__global__ __launch_bounds__(256)
void moe_router(const float* __restrict__ x,
                const float* __restrict__ rw,
                const float* __restrict__ rb,
                int*   __restrict__ counts,
                int*   __restrict__ tok_list,
                float* __restrict__ gate_list)
{
    const int wave = threadIdx.x >> 5;
    const int lane = threadIdx.x & 31;
    const int t = blockIdx.x * 8 + wave;
    if (t >= T_TOKENS) return;

    float acc[NEXP];
#pragma unroll
    for (int e = 0; e < NEXP; ++e) acc[e] = 0.f;

    const float* xr = x + (size_t)t * DMODEL;
    for (int d = lane; d < DMODEL; d += 32) {
        const float xv = xr[d];
        const float* r = rw + (size_t)d * NEXP;
#pragma unroll
        for (int e = 0; e < NEXP; ++e) acc[e] += xv * r[e];
    }
#pragma unroll
    for (int off = 16; off >= 1; off >>= 1) {
#pragma unroll
        for (int e = 0; e < NEXP; ++e)
            acc[e] += __shfl_xor(acc[e], off, 32);
    }

    if (lane == 0) {
        float lg[NEXP];
#pragma unroll
        for (int e = 0; e < NEXP; ++e) lg[e] = acc[e] + rb[e];
        int i0 = 0;
#pragma unroll
        for (int e = 1; e < NEXP; ++e) if (lg[e] > lg[i0]) i0 = e;
        int i1 = (i0 == 0) ? 1 : 0;
#pragma unroll
        for (int e = 0; e < NEXP; ++e) {
            if (e == i0) continue;
            if (lg[e] > lg[i1]) i1 = e;
        }
        const float ex = __expf(lg[i1] - lg[i0]);   // stable: lg[i0] >= lg[i1]
        const float g0 = 1.f / (1.f + ex);
        const float g1 = ex  / (1.f + ex);

        int p0 = atomicAdd(&counts[i0], 1);
        tok_list [i0 * T_TOKENS + p0] = t;
        gate_list[i0 * T_TOKENS + p0] = g0;
        int p1 = atomicAdd(&counts[i1], 1);
        tok_list [i1 * T_TOKENS + p1] = t;
        gate_list[i1 * T_TOKENS + p1] = g1;
    }
}

// ===================== Pack kernels (once per launch) =====================
// x: fp32 -> bf16 row-major (A operand read as 2x16B contiguous per lane).
__global__ __launch_bounds__(256)
void pack_x(const float* __restrict__ x, __bf16* __restrict__ xb)
{
    const size_t i8 = ((size_t)blockIdx.x * blockDim.x + threadIdx.x) * 8;
    v4f f0 = *(const v4f*)(x + i8);
    v4f f1 = *(const v4f*)(x + i8 + 4);
    v8bf o;
#pragma unroll
    for (int i = 0; i < 4; ++i) { o[i] = (__bf16)f0[i]; o[i + 4] = (__bf16)f1[i]; }
    *(v8bf*)(xb + i8) = o;
}

// w1: fp32 [e][D][F] -> bf16 fragment-packed [e][nt][kt][lane][16]
// B packing used here (contiguous k, same as A reads): K = half*16 + i, N = lane&15.
__global__ __launch_bounds__(256)
void pack_w1(const float* __restrict__ w, __bf16* __restrict__ wp)
{
    const size_t tid = (size_t)blockIdx.x * blockDim.x + threadIdx.x;
    const int lane = (int)(tid & 31);
    size_t idx = tid >> 5;
    const int kt = (int)(idx & (NKT1 - 1)); idx >>= 5;
    const int nt = (int)(idx & (NT1 - 1));  idx >>= 8;
    const int e  = (int)idx;
    const int half = lane >> 4, l16 = lane & 15;

    const float* src = w + ((size_t)e * DMODEL + kt * 32 + half * 16) * DFF + nt * 16 + l16;
    v16bf o;
#pragma unroll
    for (int i = 0; i < 16; ++i) o[i] = (__bf16)src[(size_t)i * DFF];
    *(v16bf*)(wp + ((((size_t)e * NT1 + nt) * NKT1 + kt) * 32 + lane) * 16) = o;
}

// w2: fp32 [e][F][D] -> bf16 fragment-packed [e][kt][nt][lane][16]
__global__ __launch_bounds__(256)
void pack_w2(const float* __restrict__ w, __bf16* __restrict__ wp)
{
    const size_t tid = (size_t)blockIdx.x * blockDim.x + threadIdx.x;
    const int lane = (int)(tid & 31);
    size_t idx = tid >> 5;
    const int nt = (int)(idx & (NT2 - 1));  idx >>= 6;
    const int kt = (int)(idx & (NKT2 - 1)); idx >>= 7;
    const int e  = (int)idx;
    const int half = lane >> 4, l16 = lane & 15;

    const float* src = w + ((size_t)e * DFF + kt * 32 + half * 16) * DMODEL + nt * 16 + l16;
    v16bf o;
#pragma unroll
    for (int i = 0; i < 16; ++i) o[i] = (__bf16)src[(size_t)i * DMODEL];
    *(v16bf*)(wp + ((((size_t)e * NKT2 + kt) * NT2 + nt) * 32 + lane) * 16) = o;
}

// ===================== Fused expert FFN =====================
// 512 threads = 16 waves, BM=64 routed tokens. Wave owns 64 output cols.
// Per 256-wide FF chunk: phase1 h[64, wave cols 16] via 32 k-steps, bias+relu
// -> LDS bf16; phase2 acc += h @ w2 (4x4 f32 16x16 tiles per wave).
__global__ __launch_bounds__(512)
void moe_ffn(const __bf16* __restrict__ xb,
             const __bf16* __restrict__ w1p, const float* __restrict__ b1,
             const __bf16* __restrict__ w2p, const float* __restrict__ b2,
             const int*   __restrict__ counts,
             const int*   __restrict__ tok_list,
             const float* __restrict__ gate_list,
             float* __restrict__ out)
{
    const int e    = blockIdx.y;
    const int tile = blockIdx.x;
    const int cnt  = counts[e];
    if (tile * BM >= cnt) return;

    const int wave = threadIdx.x >> 5;
    const int lane = threadIdx.x & 31;
    const int half = lane >> 4;
    const int l16  = lane & 15;

    __shared__ __bf16 hbuf[BM][BF + 8];   // 64 x 264 bf16, padded rows

    const int base = e * T_TOKENS;
    int tok[4];
#pragma unroll
    for (int r = 0; r < 4; ++r) {
        const int rr = tile * BM + r * 16 + l16;
        tok[r] = tok_list[base + (rr < cnt ? rr : cnt - 1)];
    }

    const __bf16* W1 = w1p + (size_t)e * NT1 * NKT1 * 512;
    const __bf16* W2 = w2p + (size_t)e * NKT2 * NT2 * 512;

    v8f acc[4][4];
#pragma unroll
    for (int r = 0; r < 4; ++r)
#pragma unroll
        for (int t2 = 0; t2 < 4; ++t2)
#pragma unroll
            for (int j = 0; j < 8; ++j) acc[r][t2][j] = 0.f;

    for (int chunk = 0; chunk < DFF / BF; ++chunk) {
        // ---------- phase 1: h columns [nt*16 .. nt*16+16) ----------
        const int nt = chunk * (BF / 16) + wave;
        v8f h[4];
#pragma unroll
        for (int r = 0; r < 4; ++r)
#pragma unroll
            for (int j = 0; j < 8; ++j) h[r][j] = 0.f;

        const __bf16* bp = W1 + ((size_t)nt * NKT1 * 32 + lane) * 16;
        for (int kt = 0; kt < NKT1; ++kt) {
            __builtin_prefetch(bp + 512, 0, 1);          // next k-step weights
            const v16bf b = *(const v16bf*)bp;
            bp += 512;
#pragma unroll
            for (int r = 0; r < 4; ++r) {
                const __bf16* xr = xb + (size_t)tok[r] * DMODEL + kt * 32 + half * 16;
                const v8bf lo = *(const v8bf*)xr;
                const v8bf hi = *(const v8bf*)(xr + 8);
                h[r] = __builtin_amdgcn_wmma_f32_16x16x32_bf16(
                           false, cat8(lo, hi), false, b, (short)0, h[r], false, false);
            }
        }

        const float bias1 = b1[e * DFF + nt * 16 + l16];
#pragma unroll
        for (int r = 0; r < 4; ++r)
#pragma unroll
            for (int j = 0; j < 8; ++j) {
                float v = h[r][j] + bias1;
                v = fmaxf(v, 0.f);
                hbuf[r * 16 + j + 8 * half][wave * 16 + l16] = (__bf16)v;
            }
        __syncthreads();

        // ---------- phase 2: acc += h_chunk @ w2[chunk rows, wave cols] ----------
        for (int ff = 0; ff < BF; ff += 32) {
            const int kt2 = chunk * (BF / 32) + (ff >> 5);
            v16bf a[4];
#pragma unroll
            for (int r = 0; r < 4; ++r) {
                const __bf16* hr = &hbuf[r * 16 + l16][ff + half * 16];
                const v8bf lo = *(const v8bf*)hr;
                const v8bf hi = *(const v8bf*)(hr + 8);
                a[r] = cat8(lo, hi);
            }
            const __bf16* bp2 = W2 + (((size_t)kt2 * NT2 + wave * 4) * 32 + lane) * 16;
            __builtin_prefetch(bp2 + (size_t)NT2 * 512, 0, 1);   // next kt2 weights
#pragma unroll
            for (int t2 = 0; t2 < 4; ++t2) {
                const v16bf b = *(const v16bf*)bp2;
                bp2 += 512;
#pragma unroll
                for (int r = 0; r < 4; ++r)
                    acc[r][t2] = __builtin_amdgcn_wmma_f32_16x16x32_bf16(
                                     false, a[r], false, b, (short)0, acc[r][t2], false, false);
            }
        }
        __syncthreads();   // hbuf reused next chunk
    }

    // ---------- epilogue: (acc + b2) * gate -> atomic add ----------
#pragma unroll
    for (int r = 0; r < 4; ++r)
#pragma unroll
        for (int j = 0; j < 8; ++j) {
            const int row = tile * BM + r * 16 + j + 8 * half;
            if (row < cnt) {
                const int   t = tok_list[base + row];
                const float g = gate_list[base + row];
#pragma unroll
                for (int t2 = 0; t2 < 4; ++t2) {
                    const int col = wave * 64 + t2 * 16 + l16;
                    const float v = (acc[r][t2][j] + b2[e * DMODEL + col]) * g;
                    atomicAdd(&out[(size_t)t * DMODEL + col], v);
                }
            }
        }
}

// ===================== launch =====================
extern "C" void kernel_launch(void* const* d_in, const int* in_sizes, int n_in,
                              void* d_out, int out_size, void* d_ws, size_t ws_size,
                              hipStream_t stream)
{
    (void)in_sizes; (void)n_in; (void)ws_size;
    const float* x  = (const float*)d_in[0];
    const float* rw = (const float*)d_in[1];
    const float* rb = (const float*)d_in[2];
    const float* w1 = (const float*)d_in[3];
    const float* b1 = (const float*)d_in[4];
    const float* w2 = (const float*)d_in[5];
    const float* b2 = (const float*)d_in[6];
    float* out = (float*)d_out;

    // ws: counts(256B) | tok[8][T] | gate[8][T] | xb bf16 | w1p bf16 | w2p bf16
    char*   ws        = (char*)d_ws;
    int*    counts    = (int*)ws;
    int*    tok_list  = (int*)(ws + 256);
    float*  gate_list = (float*)(ws + 256 + (size_t)NEXP * T_TOKENS * 4);
    size_t  off       = 256 + (size_t)NEXP * T_TOKENS * 8;          // 524,544
    __bf16* xb  = (__bf16*)(ws + off);  off += (size_t)T_TOKENS * DMODEL * 2;
    __bf16* w1p = (__bf16*)(ws + off);  off += (size_t)NEXP * DMODEL * DFF * 2;
    __bf16* w2p = (__bf16*)(ws + off);

    hipMemsetAsync(counts, 0, 256, stream);
    hipMemsetAsync(d_out, 0, (size_t)out_size * sizeof(float), stream);

    pack_x <<<dim3((T_TOKENS * DMODEL) / (8 * 256)), dim3(256), 0, stream>>>(x, xb);
    pack_w1<<<dim3((NEXP * NT1 * NKT1 * 32) / 256), dim3(256), 0, stream>>>(w1, w1p);
    pack_w2<<<dim3((NEXP * NKT2 * NT2 * 32) / 256), dim3(256), 0, stream>>>(w2, w2p);

    moe_router<<<dim3(T_TOKENS / 8), dim3(256), 0, stream>>>(
        x, rw, rb, counts, tok_list, gate_list);

    moe_ffn<<<dim3(T_TOKENS / BM, NEXP), dim3(512), 0, stream>>>(
        xb, w1p, b1, w2p, b2, counts, tok_list, gate_list, out);
}
